// GPT_38620345926301
// MI455X (gfx1250) — matrix operand order repaired and checked by reading, not compiled, not run on hardware
//
#include <hip/hip_runtime.h>

// ---------------------------------------------------------------------------
// Types / helpers
// ---------------------------------------------------------------------------
typedef __attribute__((ext_vector_type(16))) __bf16 v16bf;
typedef __attribute__((ext_vector_type(8)))  float  v8f;

union FragBF { unsigned u[8]; v16bf v; };

__device__ __forceinline__ unsigned short f2bf(float f) {
    unsigned u = __float_as_uint(f);
    unsigned r = u + 0x7FFFu + ((u >> 16) & 1u);   // round-to-nearest-even
    return (unsigned short)(r >> 16);
}

__device__ __forceinline__ float gelu_exact(float x) {
    return 0.5f * x * (1.0f + erff(x * 0.70710678118654752f));
}

// Async global->LDS copy of 16 bytes (CDNA5 pipe, tracked by ASYNCcnt).
// lds_off is the LDS-relative byte address (single __shared__ object => base 0).
// smem_base is passed as an (unused) asm operand so the shared array escapes:
// with the "memory" clobber the compiler must assume the asm writes it, which
// prevents the ds_load fragment reads from being folded to undef.
__device__ __forceinline__ void async_b128(char* smem_base, unsigned lds_off, const void* gaddr) {
    asm volatile("global_load_async_to_lds_b128 %0, %1, off"
                 :: "v"(lds_off), "v"(gaddr), "v"(smem_base)
                 : "memory");
}
__device__ __forceinline__ void wait_async0() {
    asm volatile("s_wait_asynccnt 0" ::: "memory");
}

// Model constants
#define Bb   2
#define Tt   827
#define Cc   1024
#define NHh  16
#define HDd  64
#define Ll   4
#define Vv   16384
#define FFf  4096
#define MR   (Bb * Tt)          // 1654 rows

// ---------------------------------------------------------------------------
// fp32 -> bf16 conversion (weights)
// ---------------------------------------------------------------------------
__global__ void cvt_kernel(const float* __restrict__ s, unsigned short* __restrict__ d, int n) {
    int i = blockIdx.x * 1024 + threadIdx.x * 4;
    #pragma unroll
    for (int j = 0; j < 4; ++j)
        if (i + j < n) d[i + j] = f2bf(s[i + j]);
}

// ---------------------------------------------------------------------------
// Adaptive bias MLP: fcout[pi*2+b, 65536] = gelu(p@A1+a1b)@A2 + a2b
// ---------------------------------------------------------------------------
__global__ void fc_kernel(const float* __restrict__ p1, const float* __restrict__ p2,
                          const float* __restrict__ p3,
                          const float* __restrict__ A1, const float* __restrict__ a1b,
                          const float* __restrict__ A2, const float* __restrict__ a2b,
                          float* __restrict__ fcout) {
    int which = blockIdx.y;                // 0..5 = pi*2 + b
    int pi = which >> 1, b = which & 1;
    const float* p = (pi == 0 ? p1 : (pi == 1 ? p2 : p3)) + b * 30;
    __shared__ float hid[60];
    int tid = threadIdx.x;
    if (tid < 60) {
        float a = a1b[tid];
        #pragma unroll
        for (int c = 0; c < 30; ++c) a += p[c] * A1[c * 60 + tid];
        hid[tid] = gelu_exact(a);
    }
    __syncthreads();
    int col = blockIdx.x * 256 + tid;
    float acc = a2b[col];
    #pragma unroll 4
    for (int kk = 0; kk < 60; ++kk) acc += hid[kk] * A2[(size_t)kk * 65536 + col];
    fcout[(size_t)which * 65536 + col] = acc;
}

// ---------------------------------------------------------------------------
// Embedding gather: x = tok_emb[idx] + pos_emb
// ---------------------------------------------------------------------------
__global__ void embed_kernel(const int* __restrict__ idx, const float* __restrict__ tok,
                             const float* __restrict__ pos, float* __restrict__ x) {
    int row = blockIdx.x;                  // 0..1653  (b*T + t)
    int t = row % Tt;
    int token = idx[row];
    const float* te = tok + (size_t)token * Cc;
    const float* pe = pos + (size_t)t * Cc;
    float* xr = x + (size_t)row * Cc;
    for (int i = threadIdx.x; i < Cc; i += 256) xr[i] = te[i] + pe[i];
}

// ---------------------------------------------------------------------------
// LayerNorm: fp32 in -> bf16 out  (one block per row)
// ---------------------------------------------------------------------------
__global__ void ln_kernel(const float* __restrict__ x, const float* __restrict__ g,
                          const float* __restrict__ bta, unsigned short* __restrict__ out) {
    int row = blockIdx.x, tid = threadIdx.x;
    __shared__ float red[256];
    const float* xr = x + (size_t)row * Cc;
    float s = 0.f;
    for (int i = tid; i < Cc; i += 256) s += xr[i];
    red[tid] = s; __syncthreads();
    for (int o = 128; o > 0; o >>= 1) { if (tid < o) red[tid] += red[tid + o]; __syncthreads(); }
    float mean = red[0] * (1.0f / Cc);
    __syncthreads();
    float vv = 0.f;
    for (int i = tid; i < Cc; i += 256) { float d = xr[i] - mean; vv += d * d; }
    red[tid] = vv; __syncthreads();
    for (int o = 128; o > 0; o >>= 1) { if (tid < o) red[tid] += red[tid + o]; __syncthreads(); }
    float rstd = rsqrtf(red[0] * (1.0f / Cc) + 1e-5f);
    for (int i = tid; i < Cc; i += 256)
        out[(size_t)row * Cc + i] = f2bf((xr[i] - mean) * rstd * g[i] + bta[i]);
}

// ---------------------------------------------------------------------------
// Tiled WMMA GEMM with double-buffered async global->LDS staging.
//   C[M,N] = A[M,K](bf16) @ B[K,N](bf16) (+bias)(+res)(gelu?)
// Block: 256 thr = 8 waves, tile 128x128, BK=32; wave -> 32x64 (2x4 WMMA tiles)
// ---------------------------------------------------------------------------
#define SA_ST 48                       // A row stride (ushorts), 96B rows (16B mult)
#define SB_ST 136                      // B row stride (ushorts), 272B rows (16B mult)
#define A_BYTES (128 * SA_ST * 2)      // 12288
#define B_BYTES (32 * SB_ST * 2)       // 8704
#define A_OFF(buf) ((unsigned)((buf) * A_BYTES))
#define B_OFF(buf) ((unsigned)(2 * A_BYTES + (buf) * B_BYTES))
#define SMEM_BYTES (2 * A_BYTES + 2 * B_BYTES)

__global__ __launch_bounds__(256) void gemm_bf16(
    const unsigned short* __restrict__ A, const unsigned short* __restrict__ Bw,
    const float* __restrict__ bias, const float* __restrict__ res,
    float* __restrict__ Cf, unsigned short* __restrict__ Cb,
    int M, int N, int K, int doGelu) {
    __shared__ __align__(16) char smem[SMEM_BYTES];
    int tid = threadIdx.x;
    int lane = tid & 31, wave = tid >> 5;
    int wm = wave >> 1, wn = wave & 1;       // 4x2 wave grid
    int l15 = lane & 15, half = lane >> 4;
    int bm = blockIdx.y * 128, bn = blockIdx.x * 128;

    // Per-thread staging assignment (same every K-step, only k0/buf change):
    // A tile: 128 rows x 32 cols; thread -> (row tid/2, 16-col half)
    int ar = tid >> 1, ac = (tid & 1) * 16;
    int agr = min(bm + ar, M - 1);           // clamp: garbage rows land in C rows >= M (never stored)
    const unsigned short* agp = A + (size_t)agr * K + ac;
    unsigned a_lds = (unsigned)((ar * SA_ST + ac) * 2);
    // B tile: 32 rows x 128 cols; thread -> (row tid/8, 16-col slice)
    int br = tid >> 3, bc = (tid & 7) * 16;
    const unsigned short* bgp = Bw + (size_t)br * N + bn + bc;
    unsigned b_lds = (unsigned)((br * SB_ST + bc) * 2);

    v8f acc[2][4];
    #pragma unroll
    for (int i = 0; i < 2; ++i)
        #pragma unroll
        for (int j = 0; j < 4; ++j) acc[i][j] = {};

    int nk = K >> 5;

    // prologue: stage tile 0 into buffer 0
    async_b128(smem, A_OFF(0) + a_lds,      agp);
    async_b128(smem, A_OFF(0) + a_lds + 16, agp + 8);
    async_b128(smem, B_OFF(0) + b_lds,      bgp);
    async_b128(smem, B_OFF(0) + b_lds + 16, bgp + 8);
    wait_async0();
    __syncthreads();

    for (int kt = 0; kt < nk; ++kt) {
        int cur = kt & 1;
        if (kt + 1 < nk) {                       // prefetch next K-tile into other buffer
            int k0n = (kt + 1) << 5;
            const unsigned short* an = agp + k0n;
            const unsigned short* bnp = bgp + (size_t)k0n * N;
            async_b128(smem, A_OFF(cur ^ 1) + a_lds,      an);
            async_b128(smem, A_OFF(cur ^ 1) + a_lds + 16, an + 8);
            async_b128(smem, B_OFF(cur ^ 1) + b_lds,      bnp);
            async_b128(smem, B_OFF(cur ^ 1) + b_lds + 16, bnp + 8);
        }

        const unsigned short* sA = (const unsigned short*)(smem + A_OFF(cur));
        const unsigned short* sB = (const unsigned short*)(smem + B_OFF(cur));

        FragBF af[2], bf[4];
        #pragma unroll
        for (int mi = 0; mi < 2; ++mi) {
            int m = wm * 32 + mi * 16 + l15;
            #pragma unroll
            for (int i = 0; i < 8; ++i) {
                int k = (i < 4) ? (half * 8 + 2 * i) : (16 + half * 8 + 2 * (i - 4));
                af[mi].u[i] = *(const unsigned*)(sA + m * SA_ST + k);
            }
        }
        #pragma unroll
        for (int ni = 0; ni < 4; ++ni) {
            int n = wn * 64 + ni * 16 + l15;
            int kb = half * 16;
            #pragma unroll
            for (int i = 0; i < 8; ++i) {
                int k = (i < 4) ? (half * 8 + 2 * i) : (16 + half * 8 + 2 * (i - 4));
                bf[ni].u[i] = (unsigned)sB[k * SB_ST + n] | ((unsigned)sB[(k + 1) * SB_ST + n] << 16);
            }
        }
        #pragma unroll
        for (int mi = 0; mi < 2; ++mi)
            #pragma unroll
            for (int ni = 0; ni < 4; ++ni)
                acc[mi][ni] = __builtin_amdgcn_wmma_f32_16x16x32_bf16(
                    false, af[mi].v, false, bf[ni].v, (short)0, acc[mi][ni], false, false);

        wait_async0();        // own async loads into next buffer done
        __syncthreads();      // everyone done reading cur + everyone's loads visible
    }

    // epilogue
    #pragma unroll
    for (int mi = 0; mi < 2; ++mi)
        #pragma unroll
        for (int ni = 0; ni < 4; ++ni) {
            int col = bn + wn * 64 + ni * 16 + l15;
            #pragma unroll
            for (int r = 0; r < 8; ++r) {
                int row = bm + wm * 32 + mi * 16 + r + 8 * half;
                if (row < M) {
                    float v = acc[mi][ni][r];
                    if (bias) v += bias[col];
                    if (res)  v += res[(size_t)row * N + col];
                    if (doGelu) v = gelu_exact(v);
                    if (Cf) Cf[(size_t)row * N + col] = v;
                    if (Cb) Cb[(size_t)row * N + col] = f2bf(v);
                }
            }
        }
}

// ---------------------------------------------------------------------------
// Region bias + multiplicative masks (per score element)
// ---------------------------------------------------------------------------
__device__ __forceinline__ float apply_regions(
    float v, int t, int s, int b, const float* __restrict__ fcout,
    const float* __restrict__ f01, const float* __restrict__ f02, const float* __restrict__ f12,
    const float* __restrict__ b01, const float* __restrict__ b02, const float* __restrict__ b12) {
    if (s < 256) {
        if (t >= 285 && t < 541) {
            int i = t - 285, j = s;
            v += fcout[(size_t)(0 * 2 + b) * 65536 + i * 256 + j];
            v *= b01[(size_t)b * 65536 + j * 256 + i] * f01[(size_t)b * 65536 + i * 256 + j];
        } else if (t >= 571 && t < 827) {
            int i = t - 571, j = s;
            v += fcout[(size_t)(1 * 2 + b) * 65536 + i * 256 + j];
            v *= b02[(size_t)b * 65536 + j * 256 + i] * f02[(size_t)b * 65536 + i * 256 + j];
        }
    } else if (s >= 286 && s < 542 && t >= 571 && t < 827) {
        int i = t - 571, j = s - 286;
        v += fcout[(size_t)(2 * 2 + b) * 65536 + i * 256 + j];
        v *= b12[(size_t)b * 65536 + j * 256 + i] * f12[(size_t)b * 65536 + i * 256 + j];
    }
    return v;
}

// ---------------------------------------------------------------------------
// Flash attention: one wave per 16 query rows; 32-key tiles so P is a
// 16x32 bf16 WMMA A-operand. Online softmax with width-16 shuffles.
// ---------------------------------------------------------------------------
__global__ __launch_bounds__(256) void attn_kernel(
    const unsigned short* __restrict__ q, const unsigned short* __restrict__ k,
    const unsigned short* __restrict__ v, unsigned short* __restrict__ y,
    const float* __restrict__ fcout,
    const float* __restrict__ f01, const float* __restrict__ f02, const float* __restrict__ f12,
    const float* __restrict__ b01, const float* __restrict__ b02, const float* __restrict__ b12) {
    __shared__ unsigned short plds[8][16][32];   // per-wave P tile (bf16)
    const int TILES = (Tt + 15) / 16;            // 52
    int tid = threadIdx.x, lane = tid & 31, w = tid >> 5;
    int flat = blockIdx.x * 8 + w;               // 0..1663
    int tile = flat % TILES;
    int bh = flat / TILES;
    int h = bh & 15, b = bh >> 4;
    int l15 = lane & 15, half = lane >> 4;
    int t0 = tile * 16;
    size_t headoff = (size_t)h * HDd;

    // Q fragments (two K=32 chunks of head dim)
    int tld = min(t0 + l15, Tt - 1);
    const unsigned short* qrow = q + ((size_t)(b * Tt + tld)) * Cc + headoff;
    FragBF qf[2];
    #pragma unroll
    for (int f = 0; f < 2; ++f)
        #pragma unroll
        for (int i = 0; i < 8; ++i) {
            int kd = f * 32 + ((i < 4) ? (half * 8 + 2 * i) : (16 + half * 8 + 2 * (i - 4)));
            qf[f].u[i] = *(const unsigned*)(qrow + kd);
        }

    float Mr[8], Ls[8];
    v8f O[4];
    #pragma unroll
    for (int r = 0; r < 8; ++r) { Mr[r] = -1e30f; Ls[r] = 0.f; }
    #pragma unroll
    for (int d = 0; d < 4; ++d) O[d] = {};

    int jmax = (t0 + 15) / 32;
    for (int j = 0; j <= jmax; ++j) {
        int sbase = j * 32;
        // K^T B-fragments: [kc][nc]  (kc = head-dim chunk, nc = key chunk of 16)
        FragBF kf[2][2];
        #pragma unroll
        for (int nc = 0; nc < 2; ++nc) {
            int skl = min(sbase + nc * 16 + l15, Tt - 1);
            const unsigned short* krow = k + ((size_t)(b * Tt + skl)) * Cc + headoff;
            #pragma unroll
            for (int kc = 0; kc < 2; ++kc)
                #pragma unroll
                for (int i = 0; i < 8; ++i) {
                    int kd = kc * 32 + ((i < 4) ? (half * 8 + 2 * i) : (16 + half * 8 + 2 * (i - 4)));
                    kf[kc][nc].u[i] = *(const unsigned*)(krow + kd);
                }
        }
        v8f s0 = {}, s1 = {};
        s0 = __builtin_amdgcn_wmma_f32_16x16x32_bf16(false, qf[0].v, false, kf[0][0].v, (short)0, s0, false, false);
        s0 = __builtin_amdgcn_wmma_f32_16x16x32_bf16(false, qf[1].v, false, kf[1][0].v, (short)0, s0, false, false);
        s1 = __builtin_amdgcn_wmma_f32_16x16x32_bf16(false, qf[0].v, false, kf[0][1].v, (short)0, s1, false, false);
        s1 = __builtin_amdgcn_wmma_f32_16x16x32_bf16(false, qf[1].v, false, kf[1][1].v, (short)0, s1, false, false);

        // bias/mask/causal + online softmax (rows r+8*half, cols = l15 / l15+16)
        #pragma unroll
        for (int r = 0; r < 8; ++r) {
            int t = t0 + r + 8 * half;
            int sc0 = sbase + l15, sc1 = sc0 + 16;
            float v0 = s0[r] * 0.125f, v1 = s1[r] * 0.125f;
            v0 = apply_regions(v0, t, sc0, b, fcout, f01, f02, f12, b01, b02, b12);
            v1 = apply_regions(v1, t, sc1, b, fcout, f01, f02, f12, b01, b02, b12);
            if (sc0 > t || t >= Tt) v0 = -1e30f;
            if (sc1 > t || t >= Tt) v1 = -1e30f;
            float mx = fmaxf(v0, v1);
            #pragma unroll
            for (int o = 8; o >= 1; o >>= 1) mx = fmaxf(mx, __shfl_xor(mx, o, 16));
            float mn = fmaxf(Mr[r], mx);
            float scale = __expf(Mr[r] - mn);
            float p0 = __expf(v0 - mn), p1 = __expf(v1 - mn);
            float rs = p0 + p1;
            #pragma unroll
            for (int o = 8; o >= 1; o >>= 1) rs += __shfl_xor(rs, o, 16);
            Ls[r] = Ls[r] * scale + rs;
            Mr[r] = mn;
            #pragma unroll
            for (int d = 0; d < 4; ++d) O[d][r] *= scale;
            plds[w][r + 8 * half][l15]      = f2bf(p0);
            plds[w][r + 8 * half][l15 + 16] = f2bf(p1);
        }
        asm volatile("s_wait_dscnt 0" ::: "memory");   // cross-lane LDS dependency, same wave

        // P as A-fragment (16x32, K = keys)
        FragBF pf;
        #pragma unroll
        for (int i = 0; i < 8; ++i) {
            int kk = (i < 4) ? (half * 8 + 2 * i) : (16 + half * 8 + 2 * (i - 4));
            pf.u[i] = *(const unsigned*)&plds[w][l15][kk];
        }
        // V B-fragments (32 keys x 64 dims), O += P @ V
        #pragma unroll
        for (int d = 0; d < 4; ++d) {
            FragBF vf;
            int nd = d * 16 + l15;
            int kb = half * 16;
            #pragma unroll
            for (int i = 0; i < 8; ++i) {
                int kq = (i < 4) ? (half * 8 + 2 * i) : (16 + half * 8 + 2 * (i - 4));
                int k0i = min(sbase + kq, Tt - 1);
                int k1i = min(sbase + kq + 1, Tt - 1);
                unsigned short a0 = v[((size_t)(b * Tt + k0i)) * Cc + headoff + nd];
                unsigned short a1 = v[((size_t)(b * Tt + k1i)) * Cc + headoff + nd];
                vf.u[i] = (unsigned)a0 | ((unsigned)a1 << 16);
            }
            O[d] = __builtin_amdgcn_wmma_f32_16x16x32_bf16(false, pf.v, false, vf.v, (short)0, O[d], false, false);
        }
    }

    // normalize + store y (bf16, [B,T,C] with head offset)
    #pragma unroll
    for (int d = 0; d < 4; ++d)
        #pragma unroll
        for (int r = 0; r < 8; ++r) {
            int t = t0 + r + 8 * half;
            if (t < Tt) {
                float val = O[d][r] / fmaxf(Ls[r], 1e-20f);
                y[((size_t)(b * Tt + t)) * Cc + headoff + d * 16 + l15] = f2bf(val);
            }
        }
}

// ---------------------------------------------------------------------------
// Host launcher
// ---------------------------------------------------------------------------
extern "C" void kernel_launch(void* const* d_in, const int* in_sizes, int n_in,
                              void* d_out, int out_size, void* d_ws, size_t ws_size,
                              hipStream_t stream) {
    const int*   idx    = (const int*)  d_in[0];
    const float* p1     = (const float*)d_in[1];
    const float* p2     = (const float*)d_in[2];
    const float* p3     = (const float*)d_in[3];
    const float* f01    = (const float*)d_in[4];
    const float* f02    = (const float*)d_in[5];
    const float* f12    = (const float*)d_in[6];
    const float* b01    = (const float*)d_in[7];
    const float* b02    = (const float*)d_in[8];
    const float* b12    = (const float*)d_in[9];
    const float* tok    = (const float*)d_in[10];
    const float* pos    = (const float*)d_in[11];
    const float* ln1_g  = (const float*)d_in[12];
    const float* ln1_b  = (const float*)d_in[13];
    const float* Wq     = (const float*)d_in[14];
    const float* bq     = (const float*)d_in[15];
    const float* Wk     = (const float*)d_in[16];
    const float* bk     = (const float*)d_in[17];
    const float* Wv     = (const float*)d_in[18];
    const float* bv     = (const float*)d_in[19];
    const float* Wp     = (const float*)d_in[20];
    const float* bp     = (const float*)d_in[21];
    const float* ln2_g  = (const float*)d_in[22];
    const float* ln2_b  = (const float*)d_in[23];
    const float* W1     = (const float*)d_in[24];
    const float* b1m    = (const float*)d_in[25];
    const float* W2     = (const float*)d_in[26];
    const float* b2m    = (const float*)d_in[27];
    const float* A1     = (const float*)d_in[28];
    const float* a1b    = (const float*)d_in[29];
    const float* A2     = (const float*)d_in[30];
    const float* a2b    = (const float*)d_in[31];
    const float* lnf_g  = (const float*)d_in[32];
    const float* lnf_b  = (const float*)d_in[33];
    const float* head_w = (const float*)d_in[34];
    const float* head_b = (const float*)d_in[35];

    // ---- workspace carve-up ----
    char* ws = (char*)d_ws;
    size_t off = 0;
    auto take = [&](size_t bytes) -> char* {
        char* p = ws + off;
        off += (bytes + 255) & ~(size_t)255;
        return p;
    };
    unsigned short* WqB = (unsigned short*)take((size_t)Ll * Cc * Cc * 2);
    unsigned short* WkB = (unsigned short*)take((size_t)Ll * Cc * Cc * 2);
    unsigned short* WvB = (unsigned short*)take((size_t)Ll * Cc * Cc * 2);
    unsigned short* WpB = (unsigned short*)take((size_t)Ll * Cc * Cc * 2);
    unsigned short* W1B = (unsigned short*)take((size_t)Ll * Cc * FFf * 2);
    unsigned short* W2B = (unsigned short*)take((size_t)Ll * FFf * Cc * 2);
    unsigned short* HWB = (unsigned short*)take((size_t)Cc * Vv * 2);
    float*          FCO = (float*)take((size_t)6 * 65536 * 4);
    float*          X   = (float*)take((size_t)MR * Cc * 4);
    unsigned short* XN  = (unsigned short*)take((size_t)MR * Cc * 2);
    unsigned short* Qb  = (unsigned short*)take((size_t)MR * Cc * 2);
    unsigned short* Kb  = (unsigned short*)take((size_t)MR * Cc * 2);
    unsigned short* Vb  = (unsigned short*)take((size_t)MR * Cc * 2);
    unsigned short* Yb  = (unsigned short*)take((size_t)MR * Cc * 2);
    unsigned short* Hb  = (unsigned short*)take((size_t)MR * FFf * 2);

    auto cvt = [&](const float* s, unsigned short* d, size_t n) {
        cvt_kernel<<<(unsigned)((n + 1023) / 1024), 256, 0, stream>>>(s, d, (int)n);
    };
    cvt(Wq, WqB, (size_t)Ll * Cc * Cc);
    cvt(Wk, WkB, (size_t)Ll * Cc * Cc);
    cvt(Wv, WvB, (size_t)Ll * Cc * Cc);
    cvt(Wp, WpB, (size_t)Ll * Cc * Cc);
    cvt(W1, W1B, (size_t)Ll * Cc * FFf);
    cvt(W2, W2B, (size_t)Ll * FFf * Cc);
    cvt(head_w, HWB, (size_t)Cc * Vv);

    fc_kernel<<<dim3(256, 6), 256, 0, stream>>>(p1, p2, p3, A1, a1b, A2, a2b, FCO);
    embed_kernel<<<MR, 256, 0, stream>>>(idx, tok, pos, X);

    const int GM = (MR + 127) / 128;   // 13
    for (int l = 0; l < Ll; ++l) {
        size_t wOff  = (size_t)l * Cc * Cc;
        size_t w1Off = (size_t)l * Cc * FFf;
        ln_kernel<<<MR, 256, 0, stream>>>(X, ln1_g + l * Cc, ln1_b + l * Cc, XN);
        gemm_bf16<<<dim3(8, GM), 256, 0, stream>>>(XN, WqB + wOff, bq + l * Cc, nullptr, nullptr, Qb, MR, Cc, Cc, 0);
        gemm_bf16<<<dim3(8, GM), 256, 0, stream>>>(XN, WkB + wOff, bk + l * Cc, nullptr, nullptr, Kb, MR, Cc, Cc, 0);
        gemm_bf16<<<dim3(8, GM), 256, 0, stream>>>(XN, WvB + wOff, bv + l * Cc, nullptr, nullptr, Vb, MR, Cc, Cc, 0);
        attn_kernel<<<208, 256, 0, stream>>>(Qb, Kb, Vb, Yb, FCO, f01, f02, f12, b01, b02, b12);
        gemm_bf16<<<dim3(8, GM), 256, 0, stream>>>(Yb, WpB + wOff, bp + l * Cc, X, X, nullptr, MR, Cc, Cc, 0);
        ln_kernel<<<MR, 256, 0, stream>>>(X, ln2_g + l * Cc, ln2_b + l * Cc, XN);
        gemm_bf16<<<dim3(32, GM), 256, 0, stream>>>(XN, W1B + w1Off, b1m + l * FFf, nullptr, nullptr, Hb, MR, FFf, Cc, 1);
        gemm_bf16<<<dim3(8, GM), 256, 0, stream>>>(Hb, W2B + w1Off, b2m + l * Cc, X, X, nullptr, MR, Cc, FFf, 0);
    }
    ln_kernel<<<MR, 256, 0, stream>>>(X, lnf_g, lnf_b, XN);
    gemm_bf16<<<dim3(128, GM), 256, 0, stream>>>(XN, HWB, head_b, nullptr, (float*)d_out, nullptr, MR, Vv, Cc, 0);
}
